// RPN_42803644072137
// MI455X (gfx1250) — matrix operand order repaired and checked by reading, not compiled, not run on hardware
//
#include <hip/hip_runtime.h>
#include <hip/hip_bf16.h>
#include <stdint.h>

// ---------------------------------------------------------------------------
// RPN pipeline for MI455X (gfx1250).
// conv3x3 as implicit GEMM on v_wmma_f32_16x16x32_f16 (compute-bound stage:
// 77.3 GFLOP vs ~45MB of traffic -> matrix cores or bust).
//  - K ordered t-major so the im2col fill is division/branch free
//  - B pre-swizzled into WMMA fragment layout -> direct global_load_b128,
//    no LDS staging for weights (L2-resident, 192MB L2)
//  - A tile double-buffered in LDS, one barrier per K-step
// Heads, ROI decode, bitonic top-k and bitmask NMS are bandwidth-trivial.
// ---------------------------------------------------------------------------

typedef __attribute__((ext_vector_type(16))) _Float16 v16h;
typedef __attribute__((ext_vector_type(8)))  _Float16 v8h;
typedef __attribute__((ext_vector_type(2)))  _Float16 v2h;
typedef __attribute__((ext_vector_type(8)))  float    v8f;

#define HF 128
#define WF 128
#define NPIX (HF * WF)          // 16384
#define CIN 512
#define COUT 512
#define KTOT (CIN * 9)          // 4608
#define NSTEP 144               // KTOT / 32
#define NANC (NPIX * 9)         // 147456
#define NSORT 262144            // next pow2 >= NANC
#define TOPN 6000
#define NKEEP 2000
#define NWORDS 188              // ceil(6000/32)

#define TILE_M 64
#define TILE_N 128
#define ASTR 36                 // padded LDS stride (halves) for bank spread

// ---- workspace layout (bytes, 256-aligned) --------------------------------
#define FEAT16_OFF 0u
#define FEAT16_SZ  (CIN * NPIX * 2u)                 // 16,777,216
#define BFRAG_OFF  (FEAT16_OFF + FEAT16_SZ)
#define BFRAG_SZ   (KTOT * COUT * 2u)                // 4,718,592
#define X16_OFF    (BFRAG_OFF + BFRAG_SZ)
#define X16_SZ     (COUT * NPIX * 2u)                // 16,777,216
#define ROIALL_OFF (X16_OFF + X16_SZ)
#define ROIALL_SZ  (NANC * 4u * 4u)                  // 2,359,296
#define KEYS_OFF   (ROIALL_OFF + ROIALL_SZ)
#define KEYS_SZ    (NSORT * 8u)                      // 2,097,152
#define BOXES_OFF  (KEYS_OFF + KEYS_SZ)
#define BOXES_SZ   (TOPN * 4u * 4u)                  // 96,000
#define KEEP_OFF   (BOXES_OFF + BOXES_SZ)
#define KEEP_SZ    (256u * 4u)
#define MASK_OFF   (KEEP_OFF + KEEP_SZ)
#define MASK_SZ    (TOPN * NWORDS * 4u)              // 4,512,000

// ---------------------------------------------------------------------------
__global__ void k_cvt_feat(const float* __restrict__ in, _Float16* __restrict__ out) {
  int i = blockIdx.x * 256 + threadIdx.x;
  if (i < CIN * NPIX) out[i] = (_Float16)in[i];
}

// Weights pre-swizzled into a 16x16x32 B-fragment layout (contiguous k per lane half, same order as A):
// Bfrag[((nt*NSTEP + ks)*32 + lane)*16 + e]  with
//   N = nt*16 + (lane&15),  K = ks*32 + (lane>>4)*16 + e,
//   K is t-major: c = K & 511, t = K >> 9,  value = conv3_w[N][c][t].
__global__ void k_prep_w(const float* __restrict__ w, _Float16* __restrict__ Bfrag) {
  int idx = blockIdx.x * 256 + threadIdx.x;
  if (idx >= KTOT * COUT) return;
  int e    = idx & 15;
  int lane = (idx >> 4) & 31;
  int tmp  = idx >> 9;
  int ks   = tmp % NSTEP;
  int nt   = tmp / NSTEP;
  int N = nt * 16 + (lane & 15);
  int K = ks * 32 + (lane >> 4) * 16 + e;
  int c = K & 511;
  int t = K >> 9;
  Bfrag[idx] = (_Float16)w[(N * CIN + c) * 9 + t];
}

// ---------------------------------------------------------------------------
// conv3x3 + bias + relu, implicit GEMM.  grid = (COUT/128, NPIX/64), block 256.
// Each wave: 32x32 C tile (4 accumulators). A via double-buffered LDS,
// B fragments straight from global (L2-resident).
__global__ __launch_bounds__(256)
void k_conv3(const _Float16* __restrict__ feat, const _Float16* __restrict__ Bfrag,
             const float* __restrict__ bias, _Float16* __restrict__ x16) {
  __shared__ _Float16 At[2][TILE_M * ASTR];

  const int tid  = threadIdx.x;
  const int lane = tid & 31;
  const int wave = tid >> 5;
  const int wm   = wave >> 2;                // 0..1
  const int wn   = wave & 3;                 // 0..3
  const int m0   = blockIdx.y * TILE_M;
  const int n0t  = blockIdx.x * (TILE_N / 16);
  const int half = lane >> 4;
  const int r    = lane & 15;

  // per-thread im2col geometry: one pixel, 8 channels (kk stride 4) per stage
  const int kk0 = tid >> 6;                  // 0..3
  const int m   = tid & 63;
  const int p   = m0 + m;
  const int ph  = p >> 7;
  const int pw  = p & 127;

  v8f acc[2][2] = {};

  auto fillA = [&](int buf, int s) {
    const int c0 = (s * 32) & 511;           // channel base (t-major K)
    const int t  = (s * 32) >> 9;            // tap 0..8, scalar for the step
    const int h  = ph + (t / 3) - 1;
    const int w  = pw + (t % 3) - 1;
    const bool inb = ((unsigned)h < 128u) & ((unsigned)w < 128u);
    const int off = ((inb ? h : 0) << 7) + (inb ? w : 0);
#pragma unroll
    for (int it = 0; it < 8; ++it) {
      int kk = kk0 + it * 4;
      _Float16 v = feat[((c0 + kk) << 14) + off];   // clamped, branchless
      if (!inb) v = (_Float16)0.f;
      At[buf][m * ASTR + kk] = v;
    }
  };

  fillA(0, 0);
  __syncthreads();

  for (int s = 0; s < NSTEP; ++s) {
    const int cur = s & 1;
    if (s + 1 < NSTEP) fillA(cur ^ 1, s + 1);
    if (s + 2 < NSTEP)
      __builtin_prefetch(&Bfrag[(((n0t + wn * 2) * NSTEP + s + 2) * 32 + lane) * 16], 0, 0);

    // A fragments (contiguous k = half*16 + e, matching Bfrag), dword pairs from LDS
    v16h afrag[2], bfrag[2];
#pragma unroll
    for (int i = 0; i < 2; ++i) {
      int row = wm * 32 + i * 16 + r;
      uint32_t* ap = (uint32_t*)&afrag[i];
#pragma unroll
      for (int v = 0; v < 8; ++v) {
        int kp = half * 16 + 2 * v;
        ap[v] = *(const uint32_t*)&At[cur][row * ASTR + kp];
      }
    }
    // B fragments: 32B per lane, contiguous per wave -> 2x global_load_b128
#pragma unroll
    for (int j = 0; j < 2; ++j) {
      int nt = n0t + wn * 2 + j;
      bfrag[j] = *(const v16h*)&Bfrag[((nt * NSTEP + s) * 32 + lane) * 16];
    }
#pragma unroll
    for (int i = 0; i < 2; ++i)
#pragma unroll
      for (int j = 0; j < 2; ++j)
        acc[i][j] = __builtin_amdgcn_wmma_f32_16x16x32_f16(
            false, afrag[i], false, bfrag[j], (short)0, acc[i][j], false, false);
    __syncthreads();
  }

  // epilogue: +bias, relu, pack 8 pixels/lane -> one b128 store
#pragma unroll
  for (int i = 0; i < 2; ++i)
#pragma unroll
    for (int j = 0; j < 2; ++j) {
      int ch = blockIdx.x * TILE_N + wn * 32 + j * 16 + r;
      float b = bias[ch];
      v8h hv;
#pragma unroll
      for (int v = 0; v < 8; ++v)
        hv[v] = (_Float16)fmaxf(acc[i][j][v] + b, 0.f);
      int pp = m0 + wm * 32 + i * 16 + half * 8;
      *(v8h*)&x16[ch * NPIX + pp] = hv;
    }
}

// ---------------------------------------------------------------------------
// 1x1 heads: grid (NPIX/512, 54), 2 pixels per thread (b32 activation loads).
__global__ __launch_bounds__(256)
void k_heads(const _Float16* __restrict__ x16,
             const float* __restrict__ cls_w, const float* __restrict__ cls_b,
             const float* __restrict__ reg_w, const float* __restrict__ reg_b,
             float* __restrict__ out_cls, float* __restrict__ out_reg) {
  int p0 = (blockIdx.x * 256 + threadIdx.x) * 2;
  int o  = blockIdx.y;
  const float* wv;
  float b;
  if (o < 18) { wv = cls_w + o * CIN;        b = cls_b[o]; }
  else        { wv = reg_w + (o - 18) * CIN; b = reg_b[o - 18]; }
  float s0 = b, s1 = b;
#pragma unroll 4
  for (int c = 0; c < CIN; ++c) {
    v2h xv = *(const v2h*)&x16[c * NPIX + p0];
    float wc = wv[c];
    s0 += (float)xv.x * wc;
    s1 += (float)xv.y * wc;
  }
  if (o < 18) {
    int k = o >> 1, t = o & 1;
    out_cls[(p0 * 9 + k) * 2 + t]       = s0;
    out_cls[((p0 + 1) * 9 + k) * 2 + t] = s1;
  } else {
    int oo = o - 18;
    int k = oo >> 2, t = oo & 3;
    out_reg[(p0 * 9 + k) * 4 + t]       = s0;
    out_reg[((p0 + 1) * 9 + k) * 4 + t] = s1;
  }
}

// ---------------------------------------------------------------------------
// anchors + make_roi + sort-key packing.  i over NSORT (pad keys with 0).
__global__ void k_roi(const float* __restrict__ out_cls, const float* __restrict__ out_reg,
                      float* __restrict__ roi_all, unsigned long long* __restrict__ keys) {
  int i = blockIdx.x * 256 + threadIdx.x;
  if (i >= NSORT) return;
  if (i >= NANC) { keys[i] = 0ull; return; }

  // anchor decode: i = ((x*128 + y)*3 + ratio)*3 + size   (x-major, per reference)
  int s  = i % 3;
  int rr = (i / 3) % 3;
  int y  = (i / 9) & 127;
  int x  = i / 1152;
  const float sqrtr[3] = {0.70710678118654752f, 1.0f, 1.41421356237309515f};
  const float sz[3]    = {8.f, 16.f, 32.f};
  float wa = 16.f * sz[s] / sqrtr[rr];
  float ha = 16.f * sz[s] * sqrtr[rr];
  float cx = (x + 0.5f) * 16.f;
  float cy = (y + 0.5f) * 16.f;
  float ax1 = rintf(cx - wa * 0.5f);   // np.round = RNE
  float ay1 = rintf(cy - ha * 0.5f);
  float ax2 = rintf(cx + wa * 0.5f);
  float ay2 = rintf(cy + ha * 0.5f);
  float aw  = ax2 - ax1, ah = ay2 - ay1;
  float acx = ax1 + aw * 0.5f, acy = ay1 + ah * 0.5f;

  float d0 = out_reg[i * 4 + 0], d1 = out_reg[i * 4 + 1];
  float d2 = out_reg[i * 4 + 2], d3 = out_reg[i * 4 + 3];
  float obj = out_cls[i * 2 + 1];

  float ccx = d0 * aw + acx;
  float ccy = d1 * ah + acy;
  float w = expf(d2) * aw;
  float h = expf(d3) * ah;
  float r0 = fmaxf(ccx - w * 0.5f, 0.f);
  float r1 = fmaxf(ccy - h * 0.5f, 0.f);
  float r2 = fmaxf(ccx + w * 0.5f, 0.f);
  float r3 = fmaxf(ccy + h * 0.5f, 0.f);
  r3 = (r3 > 2048.f) ? 2047.f : r3;
  r2 = (r2 > 2048.f) ? 2047.f : r2;
  roi_all[i * 4 + 0] = r0;
  roi_all[i * 4 + 1] = r1;
  roi_all[i * 4 + 2] = r2;
  roi_all[i * 4 + 3] = r3;

  bool valid = (r2 - r0 > 16.f) && (r3 - r1 > 16.f);
  float score = valid ? obj : -INFINITY;
  unsigned sb  = __float_as_uint(score);
  unsigned ord = (sb & 0x80000000u) ? ~sb : (sb | 0x80000000u);  // monotone map
  keys[i] = ((unsigned long long)ord << 32) | (unsigned long long)(0xFFFFFFFFu - (unsigned)i);
}

// descending bitonic pass on u64 keys (ties -> smaller anchor index first)
__global__ void k_bitonic(unsigned long long* __restrict__ keys, int j, int k) {
  int i = blockIdx.x * 256 + threadIdx.x;
  int ixj = i ^ j;
  if (ixj > i) {
    unsigned long long a = keys[i], b = keys[ixj];
    bool desc = ((i & k) == 0);
    if (desc ? (a < b) : (a > b)) { keys[i] = b; keys[ixj] = a; }
  }
}

__global__ void k_init(float* __restrict__ out_roi, unsigned* __restrict__ keep) {
  int i = blockIdx.x * 256 + threadIdx.x;
  if (i < NKEEP * 4) out_roi[i] = 0.f;
  if (i < NWORDS)    keep[i] = 0u;
}

__global__ void k_gather(const unsigned long long* __restrict__ keys,
                         const float* __restrict__ roi_all,
                         float* __restrict__ boxes, unsigned* __restrict__ keep) {
  int i = blockIdx.x * 256 + threadIdx.x;
  if (i >= TOPN) return;
  unsigned long long key = keys[i];
  unsigned idx = 0xFFFFFFFFu - (unsigned)(key & 0xFFFFFFFFull);
  unsigned hi  = (unsigned)(key >> 32);
  unsigned sb  = (hi & 0x80000000u) ? (hi ^ 0x80000000u) : ~hi;  // inverse map
  bool finite = ((sb & 0x7F800000u) != 0x7F800000u);
  float b0 = 0.f, b1 = 0.f, b2 = 0.f, b3 = 0.f;
  if (idx < NANC) {
    b0 = roi_all[idx * 4 + 0];
    b1 = roi_all[idx * 4 + 1];
    b2 = roi_all[idx * 4 + 2];
    b3 = roi_all[idx * 4 + 3];
  } else {
    finite = false;
  }
  boxes[i * 4 + 0] = b0;
  boxes[i * 4 + 1] = b1;
  boxes[i * 4 + 2] = b2;
  boxes[i * 4 + 3] = b3;
  if (finite) atomicOr(&keep[i >> 5], 1u << (i & 31));
}

// suppression bitmask: mask[i][t] bit u set iff j=t*32+u > i and iou(i,j)>=0.7
// (reference's asymmetric trunc-IoU: "g" terms from row box i, "a" from box j)
__global__ __launch_bounds__(192)
void k_mask(const float* __restrict__ boxes, unsigned* __restrict__ mask) {
  int i = blockIdx.x;
  int t = threadIdx.x;
  if (t >= NWORDS) return;
  float bi0 = boxes[i * 4 + 0], bi1 = boxes[i * 4 + 1];
  float bi2 = boxes[i * 4 + 2], bi3 = boxes[i * 4 + 3];
  float tx1 = truncf(bi0), ty1 = truncf(bi1), tx2 = truncf(bi2), ty2 = truncf(bi3);
  float gminx = fminf(tx1, tx2), gmaxx = fmaxf(tx1, tx2);
  float gminy = fminf(ty1, ty2), gmaxy = fmaxf(ty1, ty2);
  float gw = truncf(bi2 - bi0), gh = truncf(bi3 - bi1);
  float garea = gw * gh;
  unsigned bits = 0u;
#pragma unroll 4
  for (int u = 0; u < 32; ++u) {
    int j = t * 32 + u;
    if (j < TOPN && j > i) {
      float a0 = boxes[j * 4 + 0], a1 = boxes[j * 4 + 1];
      float a2 = boxes[j * 4 + 2], a3 = boxes[j * 4 + 3];
      float aw = a2 - a0, ah = a3 - a1;
      float aminx = fminf(a0, a2), amaxx = fmaxf(a0, a2);
      float aminy = fminf(a1, a3), amaxy = fmaxf(a1, a3);
      float iw = fmaxf(fminf(aminx, gminx) + aw + gw - fmaxf(amaxx, gmaxx), 0.f);
      float ih = fmaxf(fminf(aminy, gminy) + ah + gh - fmaxf(amaxy, gmaxy), 0.f);
      float inter = iw * ih;
      float uni = aw * ah + garea - inter;
      if (inter / uni >= 0.7f) bits |= (1u << u);
    }
  }
  mask[i * NWORDS + t] = bits;
}

// sequential greedy suppression (matches reference fori_loop), one workgroup
__global__ __launch_bounds__(256)
void k_nms(const unsigned* __restrict__ mask, unsigned* __restrict__ keep) {
  __shared__ unsigned kw[NWORDS];
  int t = threadIdx.x;
  if (t < NWORDS) kw[t] = keep[t];
  __syncthreads();
  for (int i = 0; i < TOPN; ++i) {
    unsigned alive = (kw[i >> 5] >> (i & 31)) & 1u;   // uniform across block
    __syncthreads();
    if (alive && t < NWORDS) kw[t] &= ~mask[i * NWORDS + t];
    __syncthreads();
  }
  if (t < NWORDS) keep[t] = kw[t];
}

__global__ void k_compact(const unsigned* __restrict__ keep,
                          const float* __restrict__ boxes,
                          float* __restrict__ out_roi) {
  if (threadIdx.x == 0 && blockIdx.x == 0) {
    int rank = 0;
    for (int i = 0; i < TOPN; ++i) {
      if ((keep[i >> 5] >> (i & 31)) & 1u) {
        if (rank < NKEEP) {
          out_roi[rank * 4 + 0] = boxes[i * 4 + 0];
          out_roi[rank * 4 + 1] = boxes[i * 4 + 1];
          out_roi[rank * 4 + 2] = boxes[i * 4 + 2];
          out_roi[rank * 4 + 3] = boxes[i * 4 + 3];
        }
        rank++;
      }
    }
  }
}

// ---------------------------------------------------------------------------
extern "C" void kernel_launch(void* const* d_in, const int* in_sizes, int n_in,
                              void* d_out, int out_size, void* d_ws, size_t ws_size,
                              hipStream_t stream) {
  const float* featmap = (const float*)d_in[0];
  const float* conv3_w = (const float*)d_in[1];
  const float* conv3_b = (const float*)d_in[2];
  const float* cls_w   = (const float*)d_in[3];
  const float* cls_b   = (const float*)d_in[4];
  const float* reg_w   = (const float*)d_in[5];
  const float* reg_b   = (const float*)d_in[6];

  float* out     = (float*)d_out;
  float* out_roi = out;                      // 2000*4
  float* out_cls = out + NKEEP * 4;          // 147456*2
  float* out_reg = out + NKEEP * 4 + NANC * 2;

  char* ws = (char*)d_ws;
  _Float16* feat16         = (_Float16*)(ws + FEAT16_OFF);
  _Float16* Bfrag          = (_Float16*)(ws + BFRAG_OFF);
  _Float16* x16            = (_Float16*)(ws + X16_OFF);
  float* roi_all           = (float*)(ws + ROIALL_OFF);
  unsigned long long* keys = (unsigned long long*)(ws + KEYS_OFF);
  float* boxes             = (float*)(ws + BOXES_OFF);
  unsigned* keep           = (unsigned*)(ws + KEEP_OFF);
  unsigned* mask           = (unsigned*)(ws + MASK_OFF);

  // prep: f32 -> f16 activations, weights -> WMMA fragment layout
  k_cvt_feat<<<(CIN * NPIX + 255) / 256, 256, 0, stream>>>(featmap, feat16);
  k_prep_w<<<(KTOT * COUT + 255) / 256, 256, 0, stream>>>(conv3_w, Bfrag);

  // conv3x3 via WMMA implicit GEMM
  k_conv3<<<dim3(COUT / TILE_N, NPIX / TILE_M), 256, 0, stream>>>(feat16, Bfrag, conv3_b, x16);

  // 1x1 heads -> pred_cls / pred_reg (written straight into d_out)
  k_heads<<<dim3(NPIX / 512, 54), 256, 0, stream>>>(x16, cls_w, cls_b, reg_w, reg_b,
                                                    out_cls, out_reg);

  // ROI decode + sort keys
  k_roi<<<NSORT / 256, 256, 0, stream>>>(out_cls, out_reg, roi_all, keys);

  // top-6000 via full descending bitonic sort (171 passes)
  for (int k = 2; k <= NSORT; k <<= 1)
    for (int j = k >> 1; j > 0; j >>= 1)
      k_bitonic<<<NSORT / 256, 256, 0, stream>>>(keys, j, k);

  // NMS
  k_init<<<(NKEEP * 4 + 255) / 256, 256, 0, stream>>>(out_roi, keep);
  k_gather<<<(TOPN + 255) / 256, 256, 0, stream>>>(keys, roi_all, boxes, keep);
  k_mask<<<TOPN, 192, 0, stream>>>(boxes, mask);
  k_nms<<<1, 256, 0, stream>>>(mask, keep);
  k_compact<<<1, 64, 0, stream>>>(keep, boxes, out_roi);
}
